// SwinTransformerBlock_21354577396038
// MI455X (gfx1250) — compile-verified, matrix operand order checked by reading, not run on hardware
//
#include <hip/hip_runtime.h>
#include <math.h>

// ---------------- CDNA5 WMMA types ----------------
typedef _Float16 half_t;
typedef __attribute__((ext_vector_type(16))) _Float16 v16h;
typedef __attribute__((ext_vector_type(8)))  float    v8f;

#define WMMA_F16(a, b, c) \
  __builtin_amdgcn_wmma_f32_16x16x32_f16(false, (a), false, (b), (short)0, (c), false, false)

// ---------------- problem constants ----------------
// B=16, L=1024, C=1024, H=W=32, WS=8, SS=4, HEADS=16, HD=64, HID=4096
// Reference quirk: MHA sequence dim = window index (S=256), batch = pos-in-window (N=64)
static constexpr int TOK  = 16384;   // B*L tokens
static constexpr int CDIM = 1024;
static constexpr int HID  = 4096;
static constexpr float EPS = 1e-5f;

// roll(-4,-4) + window partition: token (b,l) -> windowed row t = s*64 + n
__device__ __forceinline__ size_t window_row(int tok) {
  int b = tok >> 10, l = tok & 1023;
  int hh = l >> 5, ww = l & 31;
  int ho = (hh + 28) & 31, wo = (ww + 28) & 31;       // shift by -4 mod 32
  int wi = ((ho >> 3) << 2) + (wo >> 3);              // 4x4 windows
  int nn = ((ho & 7) << 3) + (wo & 7);                // pos in 8x8 window
  return ((size_t)(b * 16 + wi)) * 64 + (size_t)nn;
}

__device__ __forceinline__ float wave_sum(float v) {
  for (int o = 16; o; o >>= 1) v += __shfl_xor(v, o, 32);
  return v;
}

// ---------------- fp32 -> f16 weight conversion ----------------
__global__ __launch_bounds__(256) void cvt_f32_f16(const float* __restrict__ src,
                                                   half_t* __restrict__ dst, int n) {
  int i = blockIdx.x * 256 + threadIdx.x;
  if (i < n) dst[i] = (half_t)src[i];
}

// ---------------- LN1 + roll + window-partition (one wave per token) ----------------
__global__ __launch_bounds__(256) void ln1_window(const float* __restrict__ x,
                                                  const float* __restrict__ w,
                                                  const float* __restrict__ bia,
                                                  half_t* __restrict__ XW) {
  int lane = threadIdx.x & 31, wave = threadIdx.x >> 5;
  int tok = blockIdx.x * 8 + wave;
  const float* row = x + (size_t)tok * CDIM;
  float v[32], s = 0.f, sq = 0.f;
#pragma unroll
  for (int i = 0; i < 32; ++i) { float t = row[i * 32 + lane]; v[i] = t; s += t; sq += t * t; }
  s = wave_sum(s); sq = wave_sum(sq);
  float mean = s * (1.f / 1024.f);
  float var  = sq * (1.f / 1024.f) - mean * mean;
  float rstd = rsqrtf(var + EPS);
  half_t* dst = XW + window_row(tok) * CDIM;
#pragma unroll
  for (int i = 0; i < 32; ++i) {
    int c = i * 32 + lane;
    dst[c] = (half_t)((v[i] - mean) * rstd * w[c] + bia[c]);
  }
}

// ---------------- double-buffered WMMA mainloop ----------------
// wave tile 32(M) x 64(N); K stepped by 32, software-pipelined in register
// ping/pong buffers so WMMAs overlap the next step's global loads.
__device__ __forceinline__ void wmma_mainloop(const half_t* __restrict__ A,
                                              const half_t* __restrict__ W,
                                              size_t m0, int n0, int K,
                                              int lm, int hh, v8f (&acc)[2][4]) {
  const half_t* a0p = A + (m0 + lm) * (size_t)K + hh * 16;
  const half_t* a1p = a0p + (size_t)16 * K;
  const half_t* bp0 = W + (size_t)(n0 + 0  + lm) * K + hh * 16;
  const half_t* bp1 = W + (size_t)(n0 + 16 + lm) * K + hh * 16;
  const half_t* bp2 = W + (size_t)(n0 + 32 + lm) * K + hh * 16;
  const half_t* bp3 = W + (size_t)(n0 + 48 + lm) * K + hh * 16;

  v16h aA[2], bA[4], aB[2], bB[4];
  aA[0] = *(const v16h*)a0p;  aA[1] = *(const v16h*)a1p;
  bA[0] = *(const v16h*)bp0;  bA[1] = *(const v16h*)bp1;
  bA[2] = *(const v16h*)bp2;  bA[3] = *(const v16h*)bp3;

  int k = 0;
  for (; k + 64 < K; k += 64) {
    // stage k+32 into buffer B while computing buffer A
    aB[0] = *(const v16h*)(a0p + k + 32);  aB[1] = *(const v16h*)(a1p + k + 32);
    bB[0] = *(const v16h*)(bp0 + k + 32);  bB[1] = *(const v16h*)(bp1 + k + 32);
    bB[2] = *(const v16h*)(bp2 + k + 32);  bB[3] = *(const v16h*)(bp3 + k + 32);
    __builtin_prefetch(a0p + k + 256, 0, 1);   // global_prefetch_b8
    acc[0][0] = WMMA_F16(aA[0], bA[0], acc[0][0]);
    acc[1][0] = WMMA_F16(aA[1], bA[0], acc[1][0]);
    acc[0][1] = WMMA_F16(aA[0], bA[1], acc[0][1]);
    acc[1][1] = WMMA_F16(aA[1], bA[1], acc[1][1]);
    acc[0][2] = WMMA_F16(aA[0], bA[2], acc[0][2]);
    acc[1][2] = WMMA_F16(aA[1], bA[2], acc[1][2]);
    acc[0][3] = WMMA_F16(aA[0], bA[3], acc[0][3]);
    acc[1][3] = WMMA_F16(aA[1], bA[3], acc[1][3]);
    // stage k+64 into buffer A while computing buffer B
    aA[0] = *(const v16h*)(a0p + k + 64);  aA[1] = *(const v16h*)(a1p + k + 64);
    bA[0] = *(const v16h*)(bp0 + k + 64);  bA[1] = *(const v16h*)(bp1 + k + 64);
    bA[2] = *(const v16h*)(bp2 + k + 64);  bA[3] = *(const v16h*)(bp3 + k + 64);
    acc[0][0] = WMMA_F16(aB[0], bB[0], acc[0][0]);
    acc[1][0] = WMMA_F16(aB[1], bB[0], acc[1][0]);
    acc[0][1] = WMMA_F16(aB[0], bB[1], acc[0][1]);
    acc[1][1] = WMMA_F16(aB[1], bB[1], acc[1][1]);
    acc[0][2] = WMMA_F16(aB[0], bB[2], acc[0][2]);
    acc[1][2] = WMMA_F16(aB[1], bB[2], acc[1][2]);
    acc[0][3] = WMMA_F16(aB[0], bB[3], acc[0][3]);
    acc[1][3] = WMMA_F16(aB[1], bB[3], acc[1][3]);
  }
  // final two K-steps (k == K-64)
  aB[0] = *(const v16h*)(a0p + k + 32);  aB[1] = *(const v16h*)(a1p + k + 32);
  bB[0] = *(const v16h*)(bp0 + k + 32);  bB[1] = *(const v16h*)(bp1 + k + 32);
  bB[2] = *(const v16h*)(bp2 + k + 32);  bB[3] = *(const v16h*)(bp3 + k + 32);
#pragma unroll
  for (int j = 0; j < 4; ++j) {
    acc[0][j] = WMMA_F16(aA[0], bA[j], acc[0][j]);
    acc[1][j] = WMMA_F16(aA[1], bA[j], acc[1][j]);
  }
#pragma unroll
  for (int j = 0; j < 4; ++j) {
    acc[0][j] = WMMA_F16(aB[0], bB[j], acc[0][j]);
    acc[1][j] = WMMA_F16(aB[1], bB[j], acc[1][j]);
  }
}

// ---------------- generic GEMM: out = A(16384xK) @ W(NxK)^T + bias ----------------
enum { EP_F16 = 0, EP_F16_GELU = 1, EP_F32 = 2, EP_F32_RES = 3 };

template <int MODE>
__global__ __launch_bounds__(256) void gemm_wmma(const half_t* __restrict__ A,
                                                 const half_t* __restrict__ W,
                                                 const float* __restrict__ bias,
                                                 half_t* __restrict__ outh,
                                                 float* __restrict__ outf,
                                                 const float* __restrict__ resid,
                                                 int N, int K) {
  int lane = threadIdx.x & 31, wave = threadIdx.x >> 5;
  int lm = lane & 15, hh = lane >> 4;
  size_t m0 = (size_t)blockIdx.x * 256 + (size_t)wave * 32;
  int n0 = blockIdx.y * 64;
  v8f acc[2][4] = {};
  wmma_mainloop(A, W, m0, n0, K, lm, hh, acc);
#pragma unroll
  for (int mi = 0; mi < 2; ++mi) {
#pragma unroll
    for (int j = 0; j < 4; ++j) {
      int n = n0 + j * 16 + lm;
      float bv = bias[n];
#pragma unroll
      for (int r = 0; r < 8; ++r) {
        size_t m = m0 + mi * 16 + r + hh * 8;
        float v = acc[mi][j][r] + bv;
        size_t idx = m * (size_t)N + n;
        if (MODE == EP_F16) {
          outh[idx] = (half_t)v;
        } else if (MODE == EP_F16_GELU) {
          float g = 0.5f * v * (1.f + erff(v * 0.70710678118f));
          outh[idx] = (half_t)g;
        } else if (MODE == EP_F32) {
          outf[idx] = v;
        } else {
          outf[idx] = v + resid[idx];
        }
      }
    }
  }
}

// ---------------- QKV GEMM with scatter epilogue (V transposed on the fly) ----------------
__global__ __launch_bounds__(256) void gemm_qkv(const half_t* __restrict__ A,
                                                const half_t* __restrict__ W,
                                                const float* __restrict__ bias,
                                                half_t* __restrict__ Qb,
                                                half_t* __restrict__ Kb,
                                                half_t* __restrict__ Vt) {
  const int K = CDIM;
  int lane = threadIdx.x & 31, wave = threadIdx.x >> 5;
  int lm = lane & 15, hh = lane >> 4;
  size_t m0 = (size_t)blockIdx.x * 256 + (size_t)wave * 32;
  int n0 = blockIdx.y * 64;
  v8f acc[2][4] = {};
  wmma_mainloop(A, W, m0, n0, K, lm, hh, acc);
#pragma unroll
  for (int mi = 0; mi < 2; ++mi) {
#pragma unroll
    for (int j = 0; j < 4; ++j) {
      int col = n0 + j * 16 + lm;         // 0..3071
      int comp = col >> 10;               // 0=Q 1=K 2=V (uniform per tile)
      int cc = col & 1023;                // h*64 + d
      int h = cc >> 6, d = cc & 63;
      float bv = bias[col];
#pragma unroll
      for (int r = 0; r < 8; ++r) {
        size_t m = m0 + mi * 16 + r + hh * 8;   // token = s*64 + n
        half_t hv = (half_t)(acc[mi][j][r] + bv);
        if (comp == 0)      Qb[m * CDIM + cc] = hv;
        else if (comp == 1) Kb[m * CDIM + cc] = hv;
        else {
          int s = (int)(m >> 6), nn = (int)(m & 63);
          Vt[((size_t)((nn * 16 + h) * 64 + d)) * 256 + s] = hv;   // V^T per (n,h)
        }
      }
    }
  }
}

// ---------------- attention: one wave per (n, h, 16-query tile) ----------------
__global__ __launch_bounds__(32) void attention(const half_t* __restrict__ Qb,
                                                const half_t* __restrict__ Kb,
                                                const half_t* __restrict__ Vt,
                                                half_t* __restrict__ ATT) {
  __shared__ float sc[16][257];                    // 16x256 scores (+pad)
  __shared__ __align__(64) half_t pbuf[16][256];   // softmax probs as f16
  int lane = threadIdx.x, lm = lane & 15, hh = lane >> 4;
  int s0 = blockIdx.x * 16, h = blockIdx.y, n = blockIdx.z;

  const half_t* qp = Qb + ((size_t)((s0 + lm) * 64 + n)) * CDIM + h * 64 + hh * 16;
  v16h a0 = *(const v16h*)qp;            // d 0..31 fragment
  v16h a1 = *(const v16h*)(qp + 32);     // d 32..63 fragment

  // scores: ping/pong prefetch of key tiles
  const half_t* kp0 = Kb + ((size_t)(lm * 64 + n)) * CDIM + h * 64 + hh * 16;
  v16h b0 = *(const v16h*)kp0;
  v16h b1 = *(const v16h*)(kp0 + 32);
  for (int tt = 0; tt < 16; ++tt) {
    v16h nb0 = b0, nb1 = b1;
    if (tt < 15) {
      const half_t* kp = Kb + ((size_t)(((tt + 1) * 16 + lm) * 64 + n)) * CDIM + h * 64 + hh * 16;
      nb0 = *(const v16h*)kp;
      nb1 = *(const v16h*)(kp + 32);
    }
    v8f c = {};
    c = WMMA_F16(a0, b0, c);
    c = WMMA_F16(a1, b1, c);
#pragma unroll
    for (int r = 0; r < 8; ++r) sc[r + hh * 8][tt * 16 + lm] = c[r] * 0.125f; // HD^-0.5
    b0 = nb0; b1 = nb1;
  }
  __syncthreads();

  // softmax over 256 cols, 2 lanes per row
  int row = lane >> 1, tb = (lane & 1) * 128;
  float mx = -3.4e38f;
  for (int t = 0; t < 128; ++t) mx = fmaxf(mx, sc[row][tb + t]);
  mx = fmaxf(mx, __shfl_xor(mx, 1, 32));
  float sum = 0.f;
  for (int t = 0; t < 128; ++t) { float e = __expf(sc[row][tb + t] - mx); sc[row][tb + t] = e; sum += e; }
  sum += __shfl_xor(sum, 1, 32);
  float inv = 1.f / sum;
  for (int t = 0; t < 128; ++t) pbuf[row][tb + t] = (half_t)(sc[row][tb + t] * inv);
  __syncthreads();

  // O = P(16x256) @ V(256x64); prefetch next chunk's P (LDS) and V tiles
  v8f o[4] = {};
  size_t vbase = (size_t)((n * 16 + h) * 64);
  v16h pa = *(const v16h*)(&pbuf[lm][hh * 16]);
  v16h vb[4];
#pragma unroll
  for (int dj = 0; dj < 4; ++dj)
    vb[dj] = *(const v16h*)(Vt + (vbase + dj * 16 + lm) * 256 + hh * 16);
  for (int kk = 0; kk < 8; ++kk) {
    v16h npa = pa;
    v16h nvb[4] = {vb[0], vb[1], vb[2], vb[3]};
    if (kk < 7) {
      npa = *(const v16h*)(&pbuf[lm][(kk + 1) * 32 + hh * 16]);
#pragma unroll
      for (int dj = 0; dj < 4; ++dj)
        nvb[dj] = *(const v16h*)(Vt + (vbase + dj * 16 + lm) * 256 + (kk + 1) * 32 + hh * 16);
    }
#pragma unroll
    for (int dj = 0; dj < 4; ++dj) o[dj] = WMMA_F16(pa, vb[dj], o[dj]);
    pa = npa;
#pragma unroll
    for (int dj = 0; dj < 4; ++dj) vb[dj] = nvb[dj];
  }
#pragma unroll
  for (int dj = 0; dj < 4; ++dj)
#pragma unroll
    for (int r = 0; r < 8; ++r)
      ATT[((size_t)((s0 + r + hh * 8) * 64 + n)) * CDIM + h * 64 + dj * 16 + lm] = (half_t)o[dj][r];
}

// ---------------- window-reverse + AdaIN1 + residual + LN2 (one wave per token) ----------------
__global__ __launch_bounds__(256) void adain1_ln2(const float* __restrict__ x,
                                                  const float* __restrict__ AF,
                                                  const float* __restrict__ a_scale,
                                                  const float* __restrict__ a_shift,
                                                  const float* __restrict__ ln2w,
                                                  const float* __restrict__ ln2b,
                                                  float* __restrict__ X1,
                                                  half_t* __restrict__ H1) {
  int lane = threadIdx.x & 31, wave = threadIdx.x >> 5;
  int tok = blockIdx.x * 8 + wave;
  int l = tok & 1023;
  const float* arow = AF + window_row(tok) * CDIM;   // gather = inverse roll/partition
  float v[32], s = 0.f, sq = 0.f;
#pragma unroll
  for (int i = 0; i < 32; ++i) { float t = arow[i * 32 + lane]; v[i] = t; s += t; sq += t * t; }
  s = wave_sum(s); sq = wave_sum(sq);
  float mean = s * (1.f / 1024.f);
  float var  = (sq - 1024.f * mean * mean) * (1.f / 1023.f);   // ddof=1
  float inv  = 1.f / (sqrtf(fmaxf(var, 0.f)) + EPS);           // eps added to std
  float scl = a_scale[l], sft = a_shift[l];                    // (1,C,1) broadcast -> index by l
  const float* xrow = x + (size_t)tok * CDIM;
  float s2 = 0.f, q2 = 0.f;
#pragma unroll
  for (int i = 0; i < 32; ++i) {
    float x1 = xrow[i * 32 + lane] + (v[i] - mean) * inv * scl + sft;
    v[i] = x1; s2 += x1; q2 += x1 * x1;
  }
  s2 = wave_sum(s2); q2 = wave_sum(q2);
  float m2 = s2 * (1.f / 1024.f);
  float rstd2 = rsqrtf(q2 * (1.f / 1024.f) - m2 * m2 + EPS);
  float* x1row = X1 + (size_t)tok * CDIM;
  half_t* h1row = H1 + (size_t)tok * CDIM;
#pragma unroll
  for (int i = 0; i < 32; ++i) {
    int c = i * 32 + lane;
    x1row[c] = v[i];
    h1row[c] = (half_t)((v[i] - m2) * rstd2 * ln2w[c] + ln2b[c]);
  }
}

// ---------------- final AdaIN2 ----------------
__global__ __launch_bounds__(256) void adain2(const float* __restrict__ X2,
                                              const float* __restrict__ a_scale,
                                              const float* __restrict__ a_shift,
                                              float* __restrict__ out) {
  int lane = threadIdx.x & 31, wave = threadIdx.x >> 5;
  int tok = blockIdx.x * 8 + wave;
  int l = tok & 1023;
  const float* row = X2 + (size_t)tok * CDIM;
  float v[32], s = 0.f, sq = 0.f;
#pragma unroll
  for (int i = 0; i < 32; ++i) { float t = row[i * 32 + lane]; v[i] = t; s += t; sq += t * t; }
  s = wave_sum(s); sq = wave_sum(sq);
  float mean = s * (1.f / 1024.f);
  float var  = (sq - 1024.f * mean * mean) * (1.f / 1023.f);
  float inv  = 1.f / (sqrtf(fmaxf(var, 0.f)) + EPS);
  float scl = a_scale[l], sft = a_shift[l];
  float* orow = out + (size_t)tok * CDIM;
#pragma unroll
  for (int i = 0; i < 32; ++i) orow[i * 32 + lane] = (v[i] - mean) * inv * scl + sft;
}

// ---------------- launch ----------------
extern "C" void kernel_launch(void* const* d_in, const int* in_sizes, int n_in,
                              void* d_out, int out_size, void* d_ws, size_t ws_size,
                              hipStream_t stream) {
  (void)in_sizes; (void)n_in; (void)out_size; (void)ws_size;
  const float* x      = (const float*)d_in[0];
  const float* ln1w   = (const float*)d_in[1];
  const float* ln1b   = (const float*)d_in[2];
  const float* inw    = (const float*)d_in[3];   // (3072,1024)
  const float* inb    = (const float*)d_in[4];
  const float* outw   = (const float*)d_in[5];   // (1024,1024)
  const float* outb   = (const float*)d_in[6];
  const float* a1s    = (const float*)d_in[7];
  const float* a1sh   = (const float*)d_in[8];
  const float* ln2w   = (const float*)d_in[9];
  const float* ln2b   = (const float*)d_in[10];
  const float* w1     = (const float*)d_in[11];  // (4096,1024)
  const float* b1     = (const float*)d_in[12];
  const float* w2     = (const float*)d_in[13];  // (1024,4096)
  const float* b2     = (const float*)d_in[14];
  const float* a2s    = (const float*)d_in[15];
  const float* a2sh   = (const float*)d_in[16];

  size_t off = 0;
  char* wsb = (char*)d_ws;
  auto take = [&](size_t bytes) -> char* {
    char* p = wsb + off; off += (bytes + 255) & ~(size_t)255; return p;
  };
  half_t* WQh = (half_t*)take(3072ull * 1024 * 2);
  half_t* WOh = (half_t*)take(1024ull * 1024 * 2);
  half_t* W1h = (half_t*)take(4096ull * 1024 * 2);
  half_t* W2h = (half_t*)take(1024ull * 4096 * 2);
  half_t* XW  = (half_t*)take((size_t)TOK * CDIM * 2);   // also reused as H1 (ln2 out)
  half_t* Qb  = (half_t*)take((size_t)TOK * CDIM * 2);
  half_t* Kb  = (half_t*)take((size_t)TOK * CDIM * 2);
  half_t* Vt  = (half_t*)take((size_t)TOK * CDIM * 2);
  half_t* ATT = (half_t*)take((size_t)TOK * CDIM * 2);
  float*  AF  = (float*) take((size_t)TOK * CDIM * 4);   // out-proj result; reused as X2
  float*  X1  = (float*) take((size_t)TOK * CDIM * 4);
  half_t* G   = (half_t*)take((size_t)TOK * HID * 2);    // 16384x4096 f16

  // weights -> f16
  cvt_f32_f16<<<(3072 * 1024 + 255) / 256, 256, 0, stream>>>(inw,  WQh, 3072 * 1024);
  cvt_f32_f16<<<(1024 * 1024 + 255) / 256, 256, 0, stream>>>(outw, WOh, 1024 * 1024);
  cvt_f32_f16<<<(4096 * 1024 + 255) / 256, 256, 0, stream>>>(w1,   W1h, 4096 * 1024);
  cvt_f32_f16<<<(4096 * 1024 + 255) / 256, 256, 0, stream>>>(w2,   W2h, 4096 * 1024);

  // LN1 + shift + window partition
  ln1_window<<<TOK / 8, 256, 0, stream>>>(x, ln1w, ln1b, XW);

  // QKV projection (writes Q,K row-major + V transposed)
  gemm_qkv<<<dim3(TOK / 256, 3072 / 64), 256, 0, stream>>>(XW, WQh, inb, Qb, Kb, Vt);

  // attention: (s-tiles=16, heads=16, n=64)
  attention<<<dim3(16, 16, 64), 32, 0, stream>>>(Qb, Kb, Vt, ATT);

  // out projection -> f32 (windowed order)
  gemm_wmma<EP_F32><<<dim3(TOK / 256, 1024 / 64), 256, 0, stream>>>(
      ATT, WOh, outb, nullptr, AF, nullptr, 1024, 1024);

  // window-reverse + AdaIN1 + residual + LN2 (H1 reuses XW)
  adain1_ln2<<<TOK / 8, 256, 0, stream>>>(x, AF, a1s, a1sh, ln2w, ln2b, X1, XW);

  // MLP1 + exact GELU
  gemm_wmma<EP_F16_GELU><<<dim3(TOK / 256, HID / 64), 256, 0, stream>>>(
      XW, W1h, b1, G, nullptr, nullptr, HID, 1024);

  // MLP2 + residual -> X2 (reuses AF)
  gemm_wmma<EP_F32_RES><<<dim3(TOK / 256, 1024 / 64), 256, 0, stream>>>(
      G, W2h, b2, nullptr, AF, X1, 1024, HID);

  // final AdaIN2
  adain2<<<TOK / 8, 256, 0, stream>>>(AF, a2s, a2sh, (float*)d_out);
}
